// Network_81217831567805
// MI455X (gfx1250) — compile-verified, matrix operand order checked by reading, not run on hardware
//
#include <hip/hip_runtime.h>

typedef __attribute__((ext_vector_type(16))) _Float16 v16h;
typedef __attribute__((ext_vector_type(8)))  _Float16 v8h;
typedef __attribute__((ext_vector_type(8)))  float    v8f;

// Problem constants
#define T_STEPS 15
#define C_IN 6
#define HW_IN 160
#define F1 30
#define K1 5
#define THR1 15.0f
#define F2 250
#define THR2 10.0f
#define HP 80                 // pooled spatial
#define IPAD_W 82             // pooled + pad1
#define KDIM 270              // 30*3*3
#define KPAD 288              // padded to 9 x 32
#define MPAD 256              // 250 padded to 16 x 16
#define NSPAT 6400            // 80*80
#define PLANE (F2*NSPAT)      // 1,600,000
#define VOL (T_STEPS*PLANE)   // 24,000,000

// Workspace byte offsets (all 16B+ aligned; packed u64 is 8B aligned)
#define OFF_IPAD 0u                       // 15*30*82*82 f16 = 6,051,600 B
#define OFF_A    6051840u                 // 256*288 f16    =   147,456 B
#define OFF_WF   6199296u                 // 6400 int
#define OFF_LAST 6224896u                 // 6400 f32
#define OFF_NSPK 6250496u                 // 1.6M f32
#define OFF_VAL0 12650496u                // 1.6M f32
#define OFF_TOT  19050496u                // 1.6M f32
#define OFF_TMAX 25450496u                // u32
#define OFF_PACK 25450504u                // u64
#define OFF_ANY  25450512u                // 15*6400 int

#define IPAD_ELEMS (T_STEPS*F1*IPAD_W*IPAD_W)   // 3,025,800

// ---------------------------------------------------------------- init
__global__ void k_init(_Float16* __restrict__ ipad,
                       unsigned* __restrict__ tmax,
                       unsigned long long* __restrict__ packed)
{
    int i = blockIdx.x * blockDim.x + threadIdx.x;
    if (i < IPAD_ELEMS) ipad[i] = (_Float16)0.f;
    if (i == 0) { *tmax = 0u; *packed = 0ull; }
}

// ---------------------------------------------------------------- W2 -> f16 padded A[256][288]
__global__ void k_prep_w2(const float* __restrict__ W2, _Float16* __restrict__ A)
{
    int e = blockIdx.x * blockDim.x + threadIdx.x;   // 0..73727
    int m = e / KPAD;
    int k = e - m * KPAD;
    _Float16 v = (_Float16)0.f;
    if (m < F2 && k < KDIM) v = (_Float16)W2[m * KDIM + k];  // K index == c*9+kh*3+kw (flat)
    A[e] = v;
}

// ---------------------------------------------------------------- conv1 + fire + 2x2 maxpool (OR)
// block = (ph, t), 256 threads; writes pooled binary spikes (f16) into padded Ipad interior
__global__ __launch_bounds__(256) void k_conv1_pool(const float* __restrict__ data,
                                                    const float* __restrict__ W1,
                                                    _Float16* __restrict__ ipad)
{
    __shared__ float w_s[F1 * C_IN * K1 * K1];       // 4500 f32
    __shared__ float in_s[C_IN * 6 * 164];           // 5904 f32 (6 rows, width 164 padded)
    const int t  = blockIdx.y;
    const int ph = blockIdx.x;
    const int tid = threadIdx.x;

    for (int e = tid; e < F1 * 150; e += 256) w_s[e] = W1[e];
    for (int e = tid; e < C_IN * 6 * 164; e += 256) {
        int c   = e / (6 * 164);
        int rem = e - c * (6 * 164);
        int ry  = rem / 164;
        int x   = rem - ry * 164;
        int y   = 2 * ph + ry;                       // padded row coord
        float v = 0.f;
        if (y >= 2 && y < 162 && x >= 2 && x < 162)
            v = data[((t * C_IN + c) * HW_IN + (y - 2)) * HW_IN + (x - 2)];
        in_s[e] = v;
    }
    __syncthreads();

    for (int i = tid; i < F1 * HP; i += 256) {
        int f  = i / HP;
        int pw = i - f * HP;
        float a00 = 0.f, a01 = 0.f, a10 = 0.f, a11 = 0.f;
        const float* wf = &w_s[f * 150];
        int x0 = 2 * pw;
        for (int c = 0; c < C_IN; ++c) {
            const float* ip = &in_s[c * 6 * 164];
            for (int ki = 0; ki < K1; ++ki) {
                const float* r0 = ip + ki * 164 + x0;
                const float* r1 = r0 + 164;
                const float* wr = wf + c * 25 + ki * 5;
                #pragma unroll
                for (int kj = 0; kj < K1; ++kj) {
                    float w = wr[kj];
                    a00 += r0[kj]     * w;
                    a01 += r0[kj + 1] * w;
                    a10 += r1[kj]     * w;
                    a11 += r1[kj + 1] * w;
                }
            }
        }
        _Float16 s = (a00 > THR1 || a01 > THR1 || a10 > THR1 || a11 > THR1)
                         ? (_Float16)1.f : (_Float16)0.f;
        ipad[((t * F1 + f) * IPAD_W + (ph + 1)) * IPAD_W + (pw + 1)] = s;
    }
}

// ---------------------------------------------------------------- conv2 via WMMA f16 -> f32, + threshold
// block = (ntile 0..399, t); 256 thr = 8 waves; wave w owns M-tiles 2w, 2w+1
__global__ __launch_bounds__(256) void k_conv2_wmma(const _Float16* __restrict__ ipad,
                                                    const _Float16* __restrict__ A,
                                                    float* __restrict__ pot)
{
    __shared__ __align__(16) _Float16 Bs[16 * KPAD];   // 9216 B, [n][K]
    const int t     = blockIdx.y;
    const int ntile = blockIdx.x;
    const int h     = ntile / 5;
    const int w0    = (ntile - h * 5) * 16;
    const int tid   = threadIdx.x;

    // build im2col tile: Bs[j][k] = Ipad[t][c][h+kh][w0+j+kw], k = c*9 + kh*3 + kw
    for (int e = tid; e < 16 * KPAD; e += 256) {
        int j = e / KPAD;
        int k = e - j * KPAD;
        _Float16 v = (_Float16)0.f;
        if (k < KDIM) {
            int c  = k / 9;
            int r  = k - c * 9;
            int kh = r / 3;
            int kw = r - kh * 3;
            v = ipad[((t * F1 + c) * IPAD_W + (h + kh)) * IPAD_W + (w0 + j + kw)];
        }
        Bs[e] = v;
    }
    __syncthreads();

    const int lane = tid & 31;
    const int wave = tid >> 5;
    const int kb   = (lane & 16) ? 8 : 0;     // K sub-offset per ISA 16-bit A/B layout
    const int row  = lane & 15;               // M row for A, N col for B
    const int m0   = wave * 32;               // two 16-row M-tiles per wave

    const _Float16* Ar0 = A + (m0 + row) * KPAD;
    const _Float16* Ar1 = A + (m0 + 16 + row) * KPAD;
    const _Float16* Br  = &Bs[row * KPAD];

    v8f c0 = {}; v8f c1 = {};
    #pragma unroll
    for (int ks = 0; ks < 9; ++ks) {
        const int k0 = ks * 32 + kb;
        v8h al0 = *(const v8h*)(Ar0 + k0);
        v8h ah0 = *(const v8h*)(Ar0 + k0 + 16);
        v8h al1 = *(const v8h*)(Ar1 + k0);
        v8h ah1 = *(const v8h*)(Ar1 + k0 + 16);
        v8h bl  = *(const v8h*)(Br  + k0);
        v8h bh  = *(const v8h*)(Br  + k0 + 16);
        v16h a0, a1, b;
        #pragma unroll
        for (int i = 0; i < 8; ++i) {
            a0[i] = al0[i]; a0[i + 8] = ah0[i];
            a1[i] = al1[i]; a1[i + 8] = ah1[i];
            b[i]  = bl[i];  b[i + 8]  = bh[i];
        }
        c0 = __builtin_amdgcn_wmma_f32_16x16x32_f16(false, a0, false, b, (short)0, c0, false, false);
        c1 = __builtin_amdgcn_wmma_f32_16x16x32_f16(false, a1, false, b, (short)0, c1, false, false);
    }

    // C/D layout: VGPR r -> M = r + (lane>=16 ? 8:0), N = lane&15
    const int n    = lane & 15;
    const int madd = (lane & 16) ? 8 : 0;
    const int base = h * HP + w0 + n;
    #pragma unroll
    for (int r = 0; r < 8; ++r) {
        int m = m0 + r + madd;
        float v = c0[r]; v = (v > THR2) ? v : 0.f;
        pot[(t * F2 + m) * NSPAT + base] = v;
        int m2 = m0 + 16 + r + madd;
        float v2 = c1[r]; v2 = (v2 > THR2) ? v2 : 0.f;
        if (m2 < F2) pot[(t * F2 + m2) * NSPAT + base] = v2;
    }
}

// ---------------------------------------------------------------- inhibition metadata
// pass 1: any-feature-fired flag per (t, h, w)
__global__ __launch_bounds__(256) void k_meta_any(const float* __restrict__ pot,
                                                  int* __restrict__ anyb)
{
    int hw = blockIdx.x * blockDim.x + threadIdx.x;  // 0..6399
    int t  = blockIdx.y;
    const float* p = pot + t * PLANE + hw;
    int any = 0;
    for (int f = 0; f < F2; ++f) any |= (p[f * NSPAT] > 0.f);
    anyb[t * NSPAT + hw] = any;
}

// pass 2: first_t (ref formula), last flag, winner feature (argmax at first_t)
__global__ __launch_bounds__(256) void k_meta_final(const float* __restrict__ pot,
                                                    const int* __restrict__ anyb,
                                                    int* __restrict__ winner,
                                                    float* __restrict__ last)
{
    int hw = blockIdx.x * blockDim.x + threadIdx.x;
    int cnt = 0, lastv = 0;
    for (int t = 0; t < T_STEPS; ++t) {
        int a = anyb[t * NSPAT + hw];
        cnt += a;
        if (t == T_STEPS - 1) lastv = a;
    }
    int ft = T_STEPS - cnt;
    ft = ft < 0 ? 0 : (ft > T_STEPS - 1 ? T_STEPS - 1 : ft);
    const float* p = pot + ft * PLANE + hw;
    float best = -1.f; int bf = 0;
    for (int f = 0; f < F2; ++f) {
        float v = p[f * NSPAT];
        if (v > best) { best = v; bf = f; }   // first occurrence of max
    }
    winner[hw] = bf;
    last[hw]   = (float)lastv;
}

// ---------------------------------------------------------------- apply inhibition, spike2, k-WTA prep
__global__ __launch_bounds__(256) void k_inhibit(float* __restrict__ pot,
                                                 float* __restrict__ spike,
                                                 const int* __restrict__ winner,
                                                 const float* __restrict__ last,
                                                 float* __restrict__ nspk,
                                                 float* __restrict__ val0,
                                                 unsigned* __restrict__ tmax)
{
    __shared__ float red[256];
    int gid = blockIdx.x * blockDim.x + threadIdx.x;  // 0..1,599,999 exact
    int f  = gid / NSPAT;
    int hw = gid - f * NSPAT;
    float coef = (winner[hw] == f) ? last[hw] : 0.f;
    int nsp = 0;
    for (int t = 0; t < T_STEPS; ++t) {
        int idx = (t * F2 + f) * NSPAT + hw;
        float p = pot[idx] * coef;
        pot[idx]   = p;
        spike[idx] = (p > 0.f) ? 1.f : 0.f;
        nsp += (p > 0.f);
    }
    int ft = T_STEPS - nsp;
    ft = ft < 0 ? 0 : (ft > T_STEPS - 1 ? T_STEPS - 1 : ft);
    float v0 = pot[(ft * F2 + f) * NSPAT + hw];       // same-thread RAW, ordered
    nspk[gid] = (float)nsp;
    val0[gid] = v0;
    float trunc = (nsp > 0) ? v0 : 0.f;               // >= 0 -> uint-compare valid
    red[threadIdx.x] = trunc;
    __syncthreads();
    for (int s = 128; s > 0; s >>= 1) {
        if (threadIdx.x < s && red[threadIdx.x + s] > red[threadIdx.x])
            red[threadIdx.x] = red[threadIdx.x + s];
        __syncthreads();
    }
    if (threadIdx.x == 0) atomicMax(tmax, __float_as_uint(red[0]));
}

// ---------------------------------------------------------------- total = nspk*(val0 + 15*truncmax)
__global__ __launch_bounds__(256) void k_total(const float* __restrict__ nspk,
                                               const float* __restrict__ val0,
                                               const unsigned* __restrict__ tmax,
                                               float* __restrict__ total)
{
    int gid = blockIdx.x * blockDim.x + threadIdx.x;
    float v = (float)T_STEPS * __uint_as_float(*tmax);
    float n = nspk[gid];
    total[gid] = n * val0[gid] + n * v;
}

// ---------------------------------------------------------------- k-WTA argmax reduce (packed u64)
__global__ __launch_bounds__(256) void k_argmax(const float* __restrict__ total,
                                                unsigned long long* __restrict__ packed)
{
    __shared__ unsigned long long red[256];
    int tid  = threadIdx.x;
    int base = blockIdx.x * 2048 + tid;
    unsigned long long best = 0ull;
    #pragma unroll
    for (int i = 0; i < 8; ++i) {
        int idx = base + i * 256;
        if (idx < PLANE) {
            unsigned long long p =
                ((unsigned long long)__float_as_uint(total[idx]) << 32) | (unsigned)(~idx);
            if (p > best) best = p;           // value major, low index wins ties
        }
    }
    red[tid] = best;
    __syncthreads();
    for (int s = 128; s > 0; s >>= 1) {
        if (tid < s && red[tid + s] > red[tid]) red[tid] = red[tid + s];
        __syncthreads();
    }
    if (tid == 0) atomicMax(packed, red[0]);
}

// ---------------------------------------------------------------- k-WTA update: record winner, suppress
__global__ __launch_bounds__(256) void k_kwta_update(float* __restrict__ total,
                                                     unsigned long long* __restrict__ packed,
                                                     float* __restrict__ winners, int iter)
{
    int tid = threadIdx.x;
    unsigned long long p = *packed;
    __syncthreads();                           // everyone read p before reset
    float val    = __uint_as_float((unsigned)(p >> 32));
    unsigned idx = ~(unsigned)p;
    bool valid = (val != 0.f);
    int fi = (int)(idx / NSPAT);
    int hw = (int)(idx - (unsigned)fi * NSPAT);
    int hi = hw / HP;
    int wi = hw - hi * HP;
    if (tid == 0) {
        winners[iter * 3 + 0] = valid ? (float)fi : -1.f;
        winners[iter * 3 + 1] = valid ? (float)hi : -1.f;
        winners[iter * 3 + 2] = valid ? (float)wi : -1.f;
        *packed = 0ull;
    }
    if (valid) {
        for (int i = tid; i < NSPAT; i += 256) total[fi * NSPAT + i] = 0.f;
        for (int i = tid; i < F2 * 9; i += 256) {
            int f = i / 9;
            int d = i - f * 9;
            int dh = d / 3 - 1, dw = d - (d / 3) * 3 - 1;
            int hh = hi + dh, ww = wi + dw;
            if (hh >= 0 && hh < HP && ww >= 0 && ww < HP)
                total[f * NSPAT + hh * HP + ww] = 0.f;
        }
    }
}

// ---------------------------------------------------------------- launch
extern "C" void kernel_launch(void* const* d_in, const int* in_sizes, int n_in,
                              void* d_out, int out_size, void* d_ws, size_t ws_size,
                              hipStream_t stream)
{
    const float* data = (const float*)d_in[0];   // (15,6,160,160)
    const float* W1   = (const float*)d_in[1];   // (30,6,5,5)
    const float* W2   = (const float*)d_in[2];   // (250,30,3,3)

    float* out     = (float*)d_out;
    float* spike2  = out;                 // 24M
    float* pot2    = out + VOL;           // 24M
    float* winners = out + 2 * VOL;       // 24

    char* wsb = (char*)d_ws;
    _Float16* ipad = (_Float16*)(wsb + OFF_IPAD);
    _Float16* A    = (_Float16*)(wsb + OFF_A);
    int*   winner  = (int*)  (wsb + OFF_WF);
    float* last    = (float*)(wsb + OFF_LAST);
    float* nspk    = (float*)(wsb + OFF_NSPK);
    float* val0    = (float*)(wsb + OFF_VAL0);
    float* total   = (float*)(wsb + OFF_TOT);
    unsigned* tmax = (unsigned*)(wsb + OFF_TMAX);
    unsigned long long* packed = (unsigned long long*)(wsb + OFF_PACK);
    int* anyb      = (int*)(wsb + OFF_ANY);

    k_init<<<(IPAD_ELEMS + 255) / 256, 256, 0, stream>>>(ipad, tmax, packed);
    k_prep_w2<<<(MPAD * KPAD) / 256, 256, 0, stream>>>(W2, A);
    k_conv1_pool<<<dim3(HP, T_STEPS), 256, 0, stream>>>(data, W1, ipad);
    k_conv2_wmma<<<dim3(400, T_STEPS), 256, 0, stream>>>(ipad, A, pot2);
    k_meta_any<<<dim3(25, T_STEPS), 256, 0, stream>>>(pot2, anyb);
    k_meta_final<<<25, 256, 0, stream>>>(pot2, anyb, winner, last);
    k_inhibit<<<PLANE / 256, 256, 0, stream>>>(pot2, spike2, winner, last, nspk, val0, tmax);
    k_total<<<PLANE / 256, 256, 0, stream>>>(nspk, val0, tmax, total);
    for (int it = 0; it < 8; ++it) {
        k_argmax<<<(PLANE + 2047) / 2048, 256, 0, stream>>>(total, packed);
        k_kwta_update<<<1, 256, 0, stream>>>(total, packed, winners, it);
    }
}
